// EdgeModel_60498909331856
// MI455X (gfx1250) — matrix operand order repaired and checked by reading, not compiled, not run on hardware
//
#include <hip/hip_runtime.h>
#include <hip/hip_bf16.h>

typedef __attribute__((ext_vector_type(16))) _Float16 v16h;
typedef __attribute__((ext_vector_type(8)))  float    v8f;
typedef __attribute__((ext_vector_type(4)))  float    v4f;

// Problem constants (fixed by the reference setup)
constexpr int   B_ = 4;
constexpr int   N_ = 50000;
constexpr int   E_ = 800000;
constexpr int   F_ = 31;
constexpr float BOXSZ = 6.0f;
constexpr float HALFB = 3.0f;

constexpr int TILES_PER_B = E_ / 16;            // 50000
constexpr int TOTAL_TILES = B_ * TILES_PER_B;   // 200000
constexpr int WAVES       = 8;                  // 256 threads, wave32
constexpr int GRID        = 1000;               // 200000 / (1000*8) = 25 iters, exact

__global__ __launch_bounds__(256)
void edge_mlp_kernel(const float* __restrict__ Vf,   // [B,N,31]
                     const float* __restrict__ Vp,   // [B,N,2]
                     const int*   __restrict__ Rs,   // [B,E]
                     const int*   __restrict__ Rr,   // [B,E]
                     const float* __restrict__ W1,   // [64,64] row-major (o,i)
                     const float* __restrict__ b1,   // [64]
                     const float* __restrict__ W2,   // [64,64]
                     const float* __restrict__ b2,   // [64]
                     float*       __restrict__ out)  // [B,E,64]
{
    __shared__ alignas(16) _Float16 hbuf[WAVES][16 * 64]; // layer-1 activations (f16)
    __shared__ alignas(16) float    obuf[WAVES][16 * 64]; // output staging (f32)

    const int tid  = threadIdx.x;
    const int w    = tid >> 5;       // wave id within block
    const int lane = tid & 31;
    const int nl   = lane & 15;      // N index / M index within half-wave
    const int kh   = lane >> 4;      // K-half selector (0/1)

    // ---- Preload W1/W2 into WMMA B-layout register fragments, biases too ----
    // B-fragment element j of lane L: K = kt*32 + (L>>4)*16 + j, N = nt*16 + (L&15)
    v16h w1f[4][2], w2f[4][2];
    float bs1[4], bs2[4];
    #pragma unroll
    for (int nt = 0; nt < 4; ++nt) {
        bs1[nt] = b1[nt * 16 + nl];
        bs2[nt] = b2[nt * 16 + nl];
        #pragma unroll
        for (int kt = 0; kt < 2; ++kt) {
            const float* w1p = W1 + (size_t)(nt * 16 + nl) * 64 + kt * 32 + kh * 16;
            const float* w2p = W2 + (size_t)(nt * 16 + nl) * 64 + kt * 32 + kh * 16;
            #pragma unroll
            for (int j = 0; j < 16; ++j) {
                w1f[nt][kt][j] = (_Float16)w1p[j];
                w2f[nt][kt][j] = (_Float16)w2p[j];
            }
        }
    }

    // Uniform trip count for every block/wave: 25 iterations each.
    for (int base = blockIdx.x * WAVES; base < TOTAL_TILES; base += GRID * WAVES) {
        const int tile = base + w;                 // always < TOTAL_TILES
        const int b    = tile / TILES_PER_B;
        const int et   = tile - b * TILES_PER_B;
        const int e    = et * 16 + nl;             // this lane's edge (2 lanes/edge)

        // ---- Gather: build Efeat = [src_f(31) | rcv_f(31) | dx | dy] ----
        const int s = Rs[(size_t)b * E_ + e];
        const int r = Rr[(size_t)b * E_ + e];
        const float* ps  = Vf + ((size_t)b * N_ + s) * F_;
        const float* pr  = Vf + ((size_t)b * N_ + r) * F_;
        const float* pps = Vp + ((size_t)b * N_ + s) * 2;
        const float* ppr = Vp + ((size_t)b * N_ + r) * 2;
        float dx = pps[0] - ppr[0];
        float dy = pps[1] - ppr[1];
        dx = (dx >  HALFB) ? dx - BOXSZ : dx;  dx = (dx <= -HALFB) ? dx + BOXSZ : dx;
        dy = (dy >  HALFB) ? dy - BOXSZ : dy;  dy = (dy <= -HALFB) ? dy + BOXSZ : dy;

        // A-fragment (16x32 f16) lane layout: K = kh*16 + j
        v16h a0, a1;
        #pragma unroll
        for (int j = 0; j < 16; ++j) {
            const int K = (kh << 4) + j;                            // 0..31
            // K in [0,31): src; K==31: rcv[0]
            const float v0 = (K < F_) ? ps[K] : pr[K - F_];
            a0[j] = (_Float16)v0;
            const int K1 = K + 32;                                  // 32..63
            float v1;
            if (K1 < 62)      v1 = pr[K1 - F_];                     // rcv[1..30]
            else              v1 = (K1 == 62) ? dx : dy;
            a1[j] = (_Float16)v1;
        }

        // ---- Layer 1: h = relu(Efeat @ W1^T + b1), stash to LDS as f16 ----
        #pragma unroll
        for (int nt = 0; nt < 4; ++nt) {
            v8f c;
            #pragma unroll
            for (int q = 0; q < 8; ++q) c[q] = bs1[nt];
            c = __builtin_amdgcn_wmma_f32_16x16x32_f16(false, a0, false, w1f[nt][0],
                                                       (short)0, c, false, false);
            c = __builtin_amdgcn_wmma_f32_16x16x32_f16(false, a1, false, w1f[nt][1],
                                                       (short)0, c, false, false);
            // C/D layout: VGPR q -> M = q + kh*8, lane -> N = nt*16 + nl
            #pragma unroll
            for (int q = 0; q < 8; ++q) {
                const float v = __builtin_fmaxf(c[q], 0.0f);   // single v_max_num_f32
                hbuf[w][(q + kh * 8) * 64 + nt * 16 + nl] = (_Float16)v;
            }
        }
        __syncthreads();

        // ---- Rebuild h in A-layout from LDS (contiguous 8-half runs -> b128) ----
        v16h h0, h1;
        #pragma unroll
        for (int j = 0; j < 16; ++j) {
            const int K = (kh << 4) + j;
            h0[j] = hbuf[w][nl * 64 + K];
            h1[j] = hbuf[w][nl * 64 + K + 32];
        }

        // ---- Layer 2: out = relu(h @ W2^T + b2), stage f32 to LDS ----
        #pragma unroll
        for (int nt = 0; nt < 4; ++nt) {
            v8f c;
            #pragma unroll
            for (int q = 0; q < 8; ++q) c[q] = bs2[nt];
            c = __builtin_amdgcn_wmma_f32_16x16x32_f16(false, h0, false, w2f[nt][0],
                                                       (short)0, c, false, false);
            c = __builtin_amdgcn_wmma_f32_16x16x32_f16(false, h1, false, w2f[nt][1],
                                                       (short)0, c, false, false);
            #pragma unroll
            for (int q = 0; q < 8; ++q) {
                obuf[w][(q + kh * 8) * 64 + nt * 16 + nl] = __builtin_fmaxf(c[q], 0.0f);
            }
        }
        __syncthreads();

        // ---- Coalesced copy-out: 4 KB contiguous per tile (b128 per lane) ----
        float* op = out + (size_t)tile * 1024;   // tile*16 edges * 64 ch
        #pragma unroll
        for (int i = 0; i < 8; ++i) {
            const int idx = i * 128 + lane * 4;
            v4f v = *(const v4f*)&obuf[w][idx];
            *(v4f*)(op + idx) = v;
        }
        __syncthreads();   // protect obuf/hbuf against next iteration
    }
}

extern "C" void kernel_launch(void* const* d_in, const int* in_sizes, int n_in,
                              void* d_out, int out_size, void* d_ws, size_t ws_size,
                              hipStream_t stream) {
    const float* Vf = (const float*)d_in[0];   // V_no_pos [4,50000,31]
    const float* Vp = (const float*)d_in[1];   // V_pos    [4,50000,2]
    const int*   Rs = (const int*)  d_in[2];   // R_s      [4,800000,1]
    const int*   Rr = (const int*)  d_in[3];   // R_r      [4,800000,1]
    const float* W1 = (const float*)d_in[4];   // [64,64]
    const float* b1 = (const float*)d_in[5];   // [64]
    const float* W2 = (const float*)d_in[6];   // [64,64]
    const float* b2 = (const float*)d_in[7];   // [64]
    float* out = (float*)d_out;                // [4,800000,64]

    edge_mlp_kernel<<<GRID, 256, 0, stream>>>(Vf, Vp, Rs, Rr, W1, b1, W2, b2, out);
}
